// SDFT_11355893530762
// MI455X (gfx1250) — compile-verified, matrix operand order checked by reading, not run on hardware
//
#include <hip/hip_runtime.h>
#include <hip/hip_bf16.h>

// ---------------------------------------------------------------------------
// SDFT modulated conv for MI455X (gfx1250, wave32, WMMA bf16).
//
// Problem: B=8, C=256, H=W=128, K=3.  Dominant cost = grouped 3x3 conv
//   out[b,co,h,w] = sum_{ci,kh,kw} w_eff[b,co,ci,kh,kw] * fea[b,ci,h+kh-1,w+kw-1]
// = per-batch implicit GEMM, M=256, N=16384, Kdim=2304  (~154.6 GFLOP total).
// Compute-bound (AI ~ 550 FLOP/B) -> v_wmma_f32_16x16x32_bf16 path.
// Modulated bf16 weights (9.4 MB) live in d_ws, pre-swizzled to A-fragment
// layout; they fit entirely in the 192 MB L2.
// ---------------------------------------------------------------------------

typedef __attribute__((ext_vector_type(16))) __bf16 v16bf;
typedef __attribute__((ext_vector_type(8)))  __bf16 v8bf;
typedef __attribute__((ext_vector_type(8)))  float  v8f;

#define B_   8
#define C_   256
#define H_   128
#define W_   128
#define ND_  64
#define TAPS 9
// LDS fea stage: 3 rows x 130 x-positions (1px halo each side) x 32 ci, bf16
#define LDS_X 130

// ---------------------------------------------------------------------------
// Kernel 1: style (per-thread 64-dot) + modulate + demod (block reduction)
// + write bf16 weights pre-swizzled into WMMA A-fragment order.
//
// A fragment (16x32 bf16, wave32): lane L<16 owns row co=L, K={0..7,16..23};
// lane L+16 owns row co=L, K={8..15,24..31}; 16 halves per lane, stored as
// lane-contiguous 32B blocks -> GEMM kernel loads 2x global_load_b128/lane.
// ---------------------------------------------------------------------------
__global__ __launch_bounds__(256)
void sdft_pack(const float* __restrict__ norm_feat,  // (B, 64)
               const float* __restrict__ mod_w,      // (C, 64)
               const float* __restrict__ mod_b,      // (C,)
               const float* __restrict__ weight,     // (C, C, 3, 3)
               __bf16* __restrict__ wbf)             // swizzled A fragments
{
    const int b  = blockIdx.x >> 8;
    const int co = blockIdx.x & 255;
    const int ci = threadIdx.x;                      // 0..255 = C_in
    const float scale = 0.02083333395421505f;        // 1/sqrt(C*K*K)=1/48

    // style[b][ci] : dot over 64 dims
    float s = mod_b[ci];
    const float* nf = norm_feat + b * ND_;
    const float* mw = mod_w + ci * ND_;
#pragma unroll 8
    for (int d = 0; d < ND_; ++d) s += nf[d] * mw[d];

    // modulate 9 taps, accumulate sum of squares
    float v[TAPS];
    float ss = 0.f;
    const float* wsrc = weight + ((size_t)co * C_ + ci) * TAPS;
#pragma unroll
    for (int t = 0; t < TAPS; ++t) {
        float x = scale * wsrc[t] * s;
        v[t] = x;
        ss += x * x;
    }

    // block reduction over ci for demod
    __shared__ float red[256];
    red[ci] = ss;
    __syncthreads();
    for (int off = 128; off > 0; off >>= 1) {
        if (ci < off) red[ci] += red[ci + off];
        __syncthreads();
    }
    const float demod = rsqrtf(red[0] + 1e-8f);

    // swizzled store: fragment = (b, tap, co/16, ci/32); within fragment:
    // lane = (co&15) + ((ci&8)?16:0), half = (ci&7) + ((ci&16)?8:0)
    const int laneInFrag = (co & 15) + ((ci & 8) ? 16 : 0);
    const int halfSlot   = (ci & 7) + ((ci & 16) ? 8 : 0);
#pragma unroll
    for (int t = 0; t < TAPS; ++t) {
        const int frag = ((b * TAPS + t) * 16 + (co >> 4)) * 8 + (ci >> 5);
        wbf[(size_t)frag * 512 + laneInFrag * 16 + halfSlot] = (__bf16)(v[t] * demod);
    }
}

// ---------------------------------------------------------------------------
// Kernel 2: implicit-GEMM conv. One workgroup per (b, 128-co block, row h).
// 8 waves; wave = (waveM in 0..1) x (waveN in 0..3); wave tile = 64(co) x 32(w).
// K loop: 8 ci-chunks of 32; per chunk stage fea[3 rows][130 x][32 ci] bf16
// into LDS (transposed so B fragments are 2x ds_load_b128), then 9 taps of
// WMMA with x shifted by kw and row selected by kh.
// ---------------------------------------------------------------------------
__global__ __launch_bounds__(256)
void sdft_conv(const float* __restrict__ fea,       // (B, C, H, W)
               const __bf16* __restrict__ wbf,      // swizzled A fragments
               float* __restrict__ out)             // (B, C, H, W)
{
    const int h     = blockIdx.x;        // output row 0..127
    const int coBlk = blockIdx.y;        // 0..1 (128 co each)
    const int b     = blockIdx.z;        // 0..7
    const int tid   = threadIdx.x;
    const int lane  = tid & 31;
    const int wave  = tid >> 5;
    const int waveM = wave & 1;          // which 64-co half
    const int waveN = wave >> 1;         // which 32-px quarter

    __shared__ __align__(16) __bf16 smem[3 * LDS_X * 32];

    v8f acc[4][2];
    const v8f vzero = {0.f, 0.f, 0.f, 0.f, 0.f, 0.f, 0.f, 0.f};
#pragma unroll
    for (int m = 0; m < 4; ++m)
#pragma unroll
        for (int n = 0; n < 2; ++n) acc[m][n] = vzero;

    const float* feaB = fea + (size_t)b * C_ * H_ * W_;
    const int laneHalf16 = (lane & 16) ? 8 : 0;   // K-half owned by this lane

    for (int cc = 0; cc < 8; ++cc) {              // ci chunks of 32
        __syncthreads();                          // prev chunk fully consumed
        // ---- stage fea chunk into LDS, bf16, transposed, zero halo ----
        for (int j = tid; j < 3 * 32 * LDS_X; j += 256) {
            const int x   = j % LDS_X;            // 0..129  (w = x-1)
            const int rc  = j / LDS_X;
            const int ci  = rc & 31;
            const int row = rc >> 5;              // 0..2    (h' = h+row-1)
            const int hh  = h + row - 1;
            const int ww  = x - 1;
            float f = 0.f;
            if ((unsigned)hh < (unsigned)H_ && (unsigned)ww < (unsigned)W_)
                f = feaB[((size_t)(cc * 32 + ci) * H_ + hh) * W_ + ww];
            smem[(row * LDS_X + x) * 32 + ci] = (__bf16)f;
        }
        // prefetch next chunk's center row into L2 (speculative)
        if (cc < 7) {
            const float* pf = feaB + ((size_t)((cc + 1) * 32 + (tid & 31)) * H_ + h) * W_
                                   + ((tid >> 5) << 4);
            __builtin_prefetch(pf, 0, 1);
        }
        __syncthreads();

        // ---- 9 taps of WMMA on the staged chunk ----
#pragma unroll
        for (int tap = 0; tap < TAPS; ++tap) {
            const int kh = tap / 3;
            const int kw = tap % 3;

            // B fragments (32x16 bf16) from LDS: lane = column w, halves over ci
            v16bf bfrag[2];
#pragma unroll
            for (int n = 0; n < 2; ++n) {
                const int x    = waveN * 32 + n * 16 + (lane & 15) + kw;  // w + kw (halo)
                const int base = (kh * LDS_X + x) * 32 + laneHalf16;
                v8bf lo = *(const v8bf*)&smem[base];        // K {0..7}/{8..15}
                v8bf hi = *(const v8bf*)&smem[base + 16];   // K {16..23}/{24..31}
                bfrag[n] = __builtin_shufflevector(lo, hi, 0, 1, 2, 3, 4, 5, 6, 7,
                                                   8, 9, 10, 11, 12, 13, 14, 15);
            }
#pragma unroll
            for (int m = 0; m < 4; ++m) {
                // pre-swizzled A fragment: lane-contiguous 32B blocks
                const int frag = ((b * TAPS + tap) * 16
                                  + (coBlk * 8 + waveM * 4 + m)) * 8 + cc;
                const __bf16* ap = wbf + (size_t)frag * 512 + lane * 16;
                v8bf lo = *(const v8bf*)(ap);
                v8bf hi = *(const v8bf*)(ap + 8);
                v16bf afrag = __builtin_shufflevector(lo, hi, 0, 1, 2, 3, 4, 5, 6, 7,
                                                      8, 9, 10, 11, 12, 13, 14, 15);
#pragma unroll
                for (int n = 0; n < 2; ++n)
                    acc[m][n] = __builtin_amdgcn_wmma_f32_16x16x32_bf16(
                        false, afrag, false, bfrag[n],
                        (short)0, acc[m][n], false, false);
            }
        }
    }

    // ---- write out: C/D layout -> VGPR r holds co = base+r (lanes<16) / base+r+8 ----
    const int coBase = coBlk * 128 + waveM * 64;
    const int nBase  = waveN * 32;
#pragma unroll
    for (int m = 0; m < 4; ++m) {
#pragma unroll
        for (int n = 0; n < 2; ++n) {
            const int wcol  = nBase + n * 16 + (lane & 15);
            const int coRow = coBase + m * 16 + ((lane & 16) ? 8 : 0);
#pragma unroll
            for (int r = 0; r < 8; ++r) {
                out[(((size_t)b * C_ + (coRow + r)) * H_ + h) * W_ + wcol] = acc[m][n][r];
            }
        }
    }
}

// ---------------------------------------------------------------------------
extern "C" void kernel_launch(void* const* d_in, const int* in_sizes, int n_in,
                              void* d_out, int out_size, void* d_ws, size_t ws_size,
                              hipStream_t stream) {
    const float* fea       = (const float*)d_in[0];  // (8,256,128,128)
    const float* norm_feat = (const float*)d_in[1];  // (8,64,1,1)
    const float* mod_w     = (const float*)d_in[2];  // (256,64,1,1)
    const float* mod_b     = (const float*)d_in[3];  // (256,)
    const float* weight    = (const float*)d_in[4];  // (1,256,256,3,3)
    float* out = (float*)d_out;

    // ws: 8*9*16*8 fragments * 1024 B = 9,437,184 B of swizzled bf16 weights
    __bf16* wbf = (__bf16*)d_ws;

    sdft_pack<<<dim3(B_ * C_), dim3(256), 0, stream>>>(norm_feat, mod_w, mod_b,
                                                       weight, wbf);
    sdft_conv<<<dim3(H_, 2, B_), dim3(256), 0, stream>>>(fea, wbf, out);
}
